// CausalSelfAttention_65154653880963
// MI455X (gfx1250) — compile-verified, matrix operand order checked by reading, not run on hardware
//
#include <hip/hip_runtime.h>
#include <hip/hip_bf16.h>

// ---------------------------------------------------------------------------
// Types
// ---------------------------------------------------------------------------
typedef __bf16 bf16_t;
typedef __attribute__((ext_vector_type(16))) __bf16  bf16x16;
typedef __attribute__((ext_vector_type(8)))  float   f32x8;
typedef __attribute__((ext_vector_type(4)))  unsigned int u32x4;
typedef __attribute__((ext_vector_type(8)))  int     i32x8;
typedef __attribute__((ext_vector_type(4)))  int     i32x4;

static_assert(sizeof(bf16x16) == 32, "bf16x16 size");

// ---------------------------------------------------------------------------
// TDM support (this toolchain: 6-arg builtin). Fallback: plain b128 copies.
// ---------------------------------------------------------------------------
#ifndef USE_TDM
#if defined(__gfx1250__) && __has_builtin(__builtin_amdgcn_tensor_load_to_lds)
#define USE_TDM 1
#else
#define USE_TDM 0
#endif
#endif

#if USE_TDM
__device__ __forceinline__ unsigned lds_addr_of(const void* p) {
  // generic -> LDS address space, then to 32-bit LDS byte offset
  return (unsigned)(unsigned long long)
      (__attribute__((address_space(3))) const void*)p;
}

// 2D tile DMA: global (row-major, 2-byte elements, row stride in elements)
// -> contiguous LDS at lds_byte. Descriptor per CDNA5 ISA section 8.3/8.4.
__device__ __forceinline__ void tdm_load_2d(const void* gptr, unsigned lds_byte,
                                            unsigned tile_d0, unsigned tile_d1,
                                            unsigned long long stride_elems) {
  unsigned long long ga = (unsigned long long)gptr;
  u32x4 g0;
  g0[0] = 1u;                                     // count=1, no gather
  g0[1] = lds_byte;                               // lds_addr
  g0[2] = (unsigned)(ga & 0xFFFFFFFFu);           // global_addr[31:0]
  g0[3] = (unsigned)((ga >> 32) & 0x01FFFFFFu)    // global_addr[56:32]
          | (2u << 30);                           // type=2 (image)
  i32x8 g1;
  g1[0] = (int)(1u << 16);                        // wg_mask=0, data_size=1 (2B)
  g1[1] = (int)((tile_d0 & 0xFFFFu) << 16);       // tensor_dim0[15:0] = tile_d0
  g1[2] = (int)(((tile_d1 & 0xFFFFu) << 16)       // tensor_dim1[15:0] = tile_d1
                | ((tile_d0 >> 16) & 0xFFFFu));
  g1[3] = (int)(((tile_d0 & 0xFFFFu) << 16)       // tile_dim0
                | ((tile_d1 >> 16) & 0xFFFFu));
  g1[4] = (int)(tile_d1 & 0xFFFFu);               // tile_dim1 (tile_dim2=0)
  g1[5] = (int)(unsigned)(stride_elems & 0xFFFFFFFFull);  // dim0_stride[31:0]
  g1[6] = (int)(unsigned)((stride_elems >> 32) & 0xFFFFull); // [47:32]
  g1[7] = 0;
  i32x4 gz4 = {0, 0, 0, 0};
  i32x8 gz8 = {0, 0, 0, 0, 0, 0, 0, 0};
  __builtin_amdgcn_tensor_load_to_lds(g0, g1, gz4, gz4, gz8, 0);
}
#endif

__device__ __forceinline__ f32x8 zero8() {
  f32x8 z;
#pragma unroll
  for (int i = 0; i < 8; ++i) z[i] = 0.0f;
  return z;
}

// 16-element bf16 fragment: two b128 chunks at p and p+16 elements.
__device__ __forceinline__ bf16x16 ld_frag(const bf16_t* p) {
  union { bf16x16 v; u32x4 q[2]; } u;
  u.q[0] = *(const u32x4*)(p);
  u.q[1] = *(const u32x4*)(p + 16);
  return u.v;
}

__device__ __forceinline__ f32x8 wmma_bf16(bf16x16 a, bf16x16 b, f32x8 c) {
  return __builtin_amdgcn_wmma_f32_16x16x32_bf16(
      false, a, false, b, (short)0, c, false, false);
}

// ---------------------------------------------------------------------------
// Problem constants
// ---------------------------------------------------------------------------
#define SEQ    2048
#define BATCH  2
#define DMODEL 2048
#define NHEADS 16
#define DHEAD  128
#define NQKV   (3 * NHEADS * DHEAD)   // 6144
#define MROWS  (SEQ * BATCH)          // 4096

// ---------------------------------------------------------------------------
// fp32 -> bf16 convert (straight) and transpose-convert (KxN f32 -> NxK bf16)
// ---------------------------------------------------------------------------
__global__ void cvt_f32_to_bf16(const float* __restrict__ src,
                                bf16_t* __restrict__ dst, int n) {
  int i = blockIdx.x * 256 + threadIdx.x;
  if (i < n) dst[i] = (bf16_t)src[i];
}

__global__ void cvt_transpose_f32_to_bf16(const float* __restrict__ src,
                                          bf16_t* __restrict__ dst,
                                          int K, int N) {
  size_t idx = (size_t)blockIdx.x * 256 + threadIdx.x;
  if (idx >= (size_t)K * N) return;
  int k = (int)(idx / N), n = (int)(idx % N);
  dst[(size_t)n * K + k] = (bf16_t)src[idx];
}

// ---------------------------------------------------------------------------
// Tiled bf16 WMMA GEMM: C(MxN) = A(MxK) * BT(NxK)^T, both operands row-major
// with K contiguous, so both LDS tiles are straight 2D copies (TDM-friendly).
// Block = 128 threads (4 waves), tile 64x64, K step 32.
// EPI==0: scatter q/k into [b,h,s,d], v into vT [b,h,d,s].  EPI==1: f32 C.
// ---------------------------------------------------------------------------
template <int EPI>
__global__ __launch_bounds__(128) void gemm_bf16_kernel(
    const bf16_t* __restrict__ A, const bf16_t* __restrict__ BT,
    int M, int N, int K, float* __restrict__ Cout,
    bf16_t* __restrict__ qb, bf16_t* __restrict__ kb,
    bf16_t* __restrict__ vT) {
  __shared__ bf16_t sA[64 * 32];        // [m][k]
  __shared__ bf16_t sB[64 * 32];        // [n][k]

  const int tid  = threadIdx.x;
  const int lane = tid & 31;
  const int wv   = tid >> 5;
  const int bn   = blockIdx.x * 64;
  const int bm   = blockIdx.y * 64;

  f32x8 acc[4];
#pragma unroll
  for (int j = 0; j < 4; ++j) acc[j] = zero8();

  const int rlo  = lane & 15;
  const int koff = (lane < 16) ? 0 : 8;

  for (int kc = 0; kc < K; kc += 32) {
#if USE_TDM
    if (wv == 0) {
      tdm_load_2d(A + (size_t)bm * K + kc, lds_addr_of(sA), 32, 64,
                  (unsigned long long)K);
      tdm_load_2d(BT + (size_t)bn * K + kc, lds_addr_of(sB), 32, 64,
                  (unsigned long long)K);
      __builtin_amdgcn_s_wait_tensorcnt(0);
    }
#else
#pragma unroll
    for (int i = 0; i < 2; ++i) {
      int idx = tid + i * 128;                 // 0..255
      int row = idx >> 2;                      // 0..63
      int c8  = (idx & 3) << 3;                // 0,8,16,24
      const bf16_t* gp = A + (size_t)(bm + row) * K + kc + c8;
      *(u32x4*)(sA + row * 32 + c8) = *(const u32x4*)gp;
      if (kc + 32 < K) __builtin_prefetch(gp + 32, 0, 0);
      const bf16_t* gq = BT + (size_t)(bn + row) * K + kc + c8;
      *(u32x4*)(sB + row * 32 + c8) = *(const u32x4*)gq;
      if (kc + 32 < K) __builtin_prefetch(gq + 32, 0, 0);
    }
#endif
    __syncthreads();

    bf16x16 af = ld_frag(sA + (wv * 16 + rlo) * 32 + koff);
#pragma unroll
    for (int j = 0; j < 4; ++j) {
      bf16x16 bf = ld_frag(sB + (j * 16 + rlo) * 32 + koff);
      acc[j] = wmma_bf16(af, bf, acc[j]);
    }
    __syncthreads();
  }

  const int mbase = (lane < 16) ? 0 : 8;
#pragma unroll
  for (int j = 0; j < 4; ++j) {
#pragma unroll
    for (int r = 0; r < 8; ++r) {
      int m = bm + wv * 16 + mbase + r;
      int n = bn + j * 16 + rlo;
      float val = acc[j][r];
      if (EPI == 1) {
        Cout[(size_t)m * N + n] = val;
      } else {
        int s = m >> 1, b = m & 1;              // m = s*BATCH + b
        int which = n >> 11;                    // 0:q 1:k 2:v
        int rem = n & 2047;
        int h = rem >> 7, dd = rem & 127;
        size_t bh = (size_t)(b * NHEADS + h);
        if (which == 0)
          qb[(bh * SEQ + s) * DHEAD + dd] = (bf16_t)val;
        else if (which == 1)
          kb[(bh * SEQ + s) * DHEAD + dd] = (bf16_t)val;
        else
          vT[(bh * DHEAD + dd) * SEQ + s] = (bf16_t)val;   // transposed V
      }
    }
  }
}

// ---------------------------------------------------------------------------
// RoPE in place on q and k, layout [b,h,s,d] bf16.
// ---------------------------------------------------------------------------
__global__ void rope_kernel(bf16_t* __restrict__ qb, bf16_t* __restrict__ kb) {
  size_t idx = (size_t)blockIdx.x * 256 + threadIdx.x;
  const size_t total = (size_t)BATCH * NHEADS * SEQ * 64;
  if (idx >= total) return;
  int j = (int)(idx & 63);
  size_t rest = idx >> 6;
  int s = (int)(rest & (SEQ - 1));
  size_t bh = rest >> 11;

  float inv = __expf(-((float)(2 * j) * (1.0f / 128.0f)) * __logf(500000.0f));
  float ang = (float)s * inv;
  float c = __cosf(ang), sn = __sinf(ang);

  size_t base = (bh * SEQ + s) * DHEAD;
  float x = (float)qb[base + j], y = (float)qb[base + j + 64];
  qb[base + j]      = (bf16_t)(x * c - y * sn);
  qb[base + j + 64] = (bf16_t)(y * c + x * sn);
  x = (float)kb[base + j]; y = (float)kb[base + j + 64];
  kb[base + j]      = (bf16_t)(x * c - y * sn);
  kb[base + j + 64] = (bf16_t)(y * c + x * sn);
}

// ---------------------------------------------------------------------------
// Flash attention: one block = 4 waves = 64 query rows of one (b,h).
// 32-key tiles, online softmax, QK^T and P*V on v_wmma_f32_16x16x32_bf16.
// K tile [t][d] and V tile [d][t] are straight 2D copies (TDM-fed).
// ---------------------------------------------------------------------------
__global__ __launch_bounds__(128) void attn_kernel(
    const bf16_t* __restrict__ qb, const bf16_t* __restrict__ kb,
    const bf16_t* __restrict__ vT, const int* __restrict__ mask,
    bf16_t* __restrict__ attn) {
  __shared__ bf16_t kt[32 * 128];       // [t][d]
  __shared__ bf16_t vt[128 * 32];       // [d][t]
  __shared__ bf16_t pbuf[4][16 * 32];   // per-wave P bounce

  const int tid  = threadIdx.x;
  const int lane = tid & 31;
  const int wv   = tid >> 5;
  const int bh   = blockIdx.x;
  const int b    = bh >> 4;
  const int h    = bh & 15;
  const int qs0  = blockIdx.y * 64 + wv * 16;

  const int rlo   = lane & 15;
  const int koff  = (lane < 16) ? 0 : 8;
  const int mbase = (lane < 16) ? 0 : 8;
  const size_t headoff = (size_t)bh * SEQ * DHEAD;

  bf16x16 aq[4];
  {
    const bf16_t* qrow = qb + headoff + (size_t)(qs0 + rlo) * DHEAD;
#pragma unroll
    for (int kc = 0; kc < 4; ++kc) aq[kc] = ld_frag(qrow + kc * 32 + koff);
  }

  float mrun[8], lrun[8];
#pragma unroll
  for (int r = 0; r < 8; ++r) { mrun[r] = -1e30f; lrun[r] = 0.0f; }
  f32x8 o[8];
#pragma unroll
  for (int j = 0; j < 8; ++j) o[j] = zero8();

  const float scale = 0.088388347648318447f;   // 1/sqrt(128)

  for (int t0 = 0; t0 < SEQ; t0 += 32) {
#if USE_TDM
    if (wv == 0) {
      tdm_load_2d(kb + headoff + (size_t)t0 * DHEAD, lds_addr_of(kt),
                  128, 32, (unsigned long long)DHEAD);
      tdm_load_2d(vT + (size_t)bh * DHEAD * SEQ + t0, lds_addr_of(vt),
                  32, 128, (unsigned long long)SEQ);
      __builtin_amdgcn_s_wait_tensorcnt(0);
    }
#else
#pragma unroll
    for (int i = 0; i < 4; ++i) {
      int idx = tid + i * 128;                 // 0..511
      int tr = idx >> 4;                       // 0..31
      int c8 = (idx & 15) << 3;                // 0..120
      *(u32x4*)(kt + tr * 128 + c8) =
          *(const u32x4*)(kb + headoff + (size_t)(t0 + tr) * DHEAD + c8);
      int dr = idx >> 2;                       // 0..127
      int t8 = (idx & 3) << 3;                 // 0,8,16,24
      *(u32x4*)(vt + dr * 32 + t8) =
          *(const u32x4*)(vT + ((size_t)bh * DHEAD + dr) * SEQ + t0 + t8);
    }
#endif
    __syncthreads();

    // ---- S = Q * K^T ----
    f32x8 s0 = zero8(), s1 = zero8();
#pragma unroll
    for (int kc = 0; kc < 4; ++kc) {
      bf16x16 b0 = ld_frag(kt + rlo * 128 + kc * 32 + koff);
      bf16x16 b1 = ld_frag(kt + (16 + rlo) * 128 + kc * 32 + koff);
      s0 = wmma_bf16(aq[kc], b0, s0);
      s1 = wmma_bf16(aq[kc], b1, s1);
    }

    float mk0 = (mask[b * SEQ + t0 + rlo] != 0) ? 0.0f : -1e30f;
    float mk1 = (mask[b * SEQ + t0 + 16 + rlo] != 0) ? 0.0f : -1e30f;

    bf16_t* pw = pbuf[wv];
#pragma unroll
    for (int r = 0; r < 8; ++r) {
      float v0 = s0[r] * scale + mk0;
      float v1 = s1[r] * scale + mk1;
      float rmax = fmaxf(v0, v1);
#pragma unroll
      for (int st = 1; st < 16; st <<= 1)
        rmax = fmaxf(rmax, __shfl_xor(rmax, st, 32));
      float mn = fmaxf(mrun[r], rmax);
      float alpha = __expf(mrun[r] - mn);
      float p0 = __expf(v0 - mn);
      float p1 = __expf(v1 - mn);
      float rs = p0 + p1;
#pragma unroll
      for (int st = 1; st < 16; st <<= 1)
        rs += __shfl_xor(rs, st, 32);
      mrun[r] = mn;
      lrun[r] = lrun[r] * alpha + rs;
#pragma unroll
      for (int jt = 0; jt < 8; ++jt) o[jt][r] *= alpha;
      int prow = mbase + r;
      pw[prow * 32 + rlo]      = (bf16_t)p0;
      pw[prow * 32 + 16 + rlo] = (bf16_t)p1;
    }
    __asm__ volatile("s_wait_dscnt 0" ::: "memory");

    // ---- O += P * V ----
    bf16x16 pa = ld_frag(pw + rlo * 32 + koff);
#pragma unroll
    for (int jt = 0; jt < 8; ++jt) {
      bf16x16 bv = ld_frag(vt + (jt * 16 + rlo) * 32 + koff);
      o[jt] = wmma_bf16(pa, bv, o[jt]);
    }
    __syncthreads();
  }

#pragma unroll
  for (int jt = 0; jt < 8; ++jt) {
#pragma unroll
    for (int r = 0; r < 8; ++r) {
      int srow = qs0 + mbase + r;
      int dd = jt * 16 + rlo;
      float val = o[jt][r] / lrun[r];
      attn[((size_t)(srow * BATCH + b)) * (NHEADS * DHEAD) + h * DHEAD + dd] =
          (bf16_t)val;
    }
  }
}

// ---------------------------------------------------------------------------
// Host launch
// ---------------------------------------------------------------------------
extern "C" void kernel_launch(void* const* d_in, const int* in_sizes, int n_in,
                              void* d_out, int out_size, void* d_ws,
                              size_t ws_size, hipStream_t stream) {
  const float* hidden = (const float*)d_in[0];   // (S,B,D) f32
  const int*   smask  = (const int*)d_in[1];     // (B,S) i32
  const float* Wqkv   = (const float*)d_in[2];   // (D, 6144) f32
  const float* Wo     = (const float*)d_in[3];   // (2048, 2048) f32
  float* out = (float*)d_out;                    // (S,B,2048) f32

  char* ws = (char*)d_ws;
  const size_t nHA   = (size_t)MROWS * DMODEL;               // 8388608
  const size_t nWqkv = (size_t)DMODEL * NQKV;                // 12582912
  const size_t nWo   = (size_t)DMODEL * DMODEL;              // 4194304
  const size_t nHead = (size_t)BATCH * NHEADS * SEQ * DHEAD; // 8388608

  bf16_t* hA     = (bf16_t*)(ws);
  bf16_t* wqkvT  = (bf16_t*)(ws + nHA * 2);
  bf16_t* woT    = (bf16_t*)(ws + (nHA + nWqkv) * 2);
  bf16_t* qbuf   = (bf16_t*)(ws + (nHA + nWqkv + nWo) * 2);
  bf16_t* kbuf   = qbuf + nHead;
  bf16_t* vTbuf  = kbuf + nHead;
  bf16_t* attnb  = vTbuf + nHead;

  // 1) bf16 conversions; weights transposed so GEMM B tiles are K-contiguous
  cvt_f32_to_bf16<<<(int)((nHA + 255) / 256), 256, 0, stream>>>(hidden, hA,
                                                                (int)nHA);
  cvt_transpose_f32_to_bf16<<<(int)((nWqkv + 255) / 256), 256, 0, stream>>>(
      Wqkv, wqkvT, DMODEL, NQKV);
  cvt_transpose_f32_to_bf16<<<(int)((nWo + 255) / 256), 256, 0, stream>>>(
      Wo, woT, DMODEL, DMODEL);

  // 2) QKV projection; q/k -> [b,h,s,d], v -> transposed [b,h,d,s]
  gemm_bf16_kernel<0><<<dim3(NQKV / 64, MROWS / 64), 128, 0, stream>>>(
      hA, wqkvT, MROWS, NQKV, DMODEL, nullptr, qbuf, kbuf, vTbuf);

  // 3) RoPE in place on q,k
  {
    size_t total = (size_t)BATCH * NHEADS * SEQ * 64;
    rope_kernel<<<(int)((total + 255) / 256), 256, 0, stream>>>(qbuf, kbuf);
  }

  // 4) flash attention
  attn_kernel<<<dim3(BATCH * NHEADS, SEQ / 64), 128, 0, stream>>>(
      qbuf, kbuf, vTbuf, smask, attnb);

  // 5) output projection -> f32 d_out
  gemm_bf16_kernel<1><<<dim3(DMODEL / 64, MROWS / 64), 128, 0, stream>>>(
      attnb, woT, MROWS, DMODEL, DMODEL, out, nullptr, nullptr, nullptr);
}
